// MultiHeadAttention_17763984736329
// MI455X (gfx1250) — matrix operand order repaired and checked by reading, not compiled, not run on hardware
//
#include <hip/hip_runtime.h>
#include <hip/hip_bf16.h>

typedef __attribute__((ext_vector_type(16))) _Float16 v16h;
typedef __attribute__((ext_vector_type(8)))  float    v8f;

#define L_SEQ 2048
#define DM    1024
#define NH    16
#define DK    64
#define BB    2
#define BH    (BB * NH)        // 32
#define M_TOT (BB * L_SEQ)     // 4096

// ---------------------------------------------------------------------------
// Fragment loaders (wave32); A and B share a contiguous k order, not AMD's table order:
//  A (16x32 f16, MxK): lane.M = lane%16, g = lane/16; half j -> K = 16*g + j
//  B (32x16 f16, KxN): lane.N = lane%16, g = lane/16; half j -> K = 16*g + j
//  C/D (16x16 f32):    lane.N = lane%16; VGPR r -> M = r + 8*(lane/16)
// ---------------------------------------------------------------------------
__device__ __forceinline__ v16h frag_a_f32(const float* __restrict__ row, int g) {
  v16h a;
  const float* p = row + 16 * g;
#pragma unroll
  for (int j = 0; j < 16; ++j) a[j] = (_Float16)p[j];
  return a;
}
__device__ __forceinline__ v16h frag_a_f16(const _Float16* row, int g) {
  v16h a;
  const _Float16* p = row + 16 * g;
#pragma unroll
  for (int j = 0; j < 16; ++j) a[j] = p[j];
  return a;
}
// B fragment when logical B[k][n] = M[n][k] for row-major M (contiguous in k).
__device__ __forceinline__ v16h frag_bt_f32(const float* __restrict__ row, int g) {
  v16h b;
  const float* p = row + 16 * g;
#pragma unroll
  for (int j = 0; j < 16; ++j) b[j] = (_Float16)p[j];
  return b;
}
__device__ __forceinline__ v16h frag_bt_f16(const _Float16* row, int g) {
  v16h b;
  const _Float16* p = row + 16 * g;
#pragma unroll
  for (int j = 0; j < 16; ++j) b[j] = p[j];
  return b;
}
__device__ __forceinline__ v8f wmma16(v16h a, v16h b, v8f c) {
  return __builtin_amdgcn_wmma_f32_16x16x32_f16(false, a, false, b, (short)0, c,
                                                false, false);
}

// ---------------------------------------------------------------------------
// Kernel 1: fused QKV projection.  y = X @ W^T.
//   Q, K written f16 as (B,H,L,DK); V written f16 TRANSPOSED as (B,H,DK,L)
//   so the attention ctx GEMM reads V with contiguous 32-byte fragments.
// ---------------------------------------------------------------------------
__global__ void k_qkv_proj(const float* __restrict__ q_in, const float* __restrict__ k_in,
                           const float* __restrict__ v_in,
                           const float* __restrict__ Wq, const float* __restrict__ Wk,
                           const float* __restrict__ Wv,
                           _Float16* __restrict__ Qh, _Float16* __restrict__ Kh,
                           _Float16* __restrict__ Vt) {
  const int lane = threadIdx.x & 31, lm = lane & 15, g = lane >> 4;
  const int wave = threadIdx.x >> 5;
  const int tile = blockIdx.x * 8 + wave;     // 49152 tiles total
  const int tm = tile & 255;                  // 256 M tiles
  const int tn = tile >> 8;                   // 192 N tiles (3 projections x 64)
  const int p  = tn >> 6;
  const int o0 = (tn & 63) * 16;
  const float* X = (p == 0) ? q_in : (p == 1) ? k_in : v_in;
  const float* W = (p == 0) ? Wq   : (p == 1) ? Wk   : Wv;
  const int mbase = tm * 16;
  const float* arow = X + (size_t)(mbase + lm) * DM;
  const float* brow = W + (size_t)(o0 + lm) * DM;
  v8f c = {};
#pragma unroll 4
  for (int kb = 0; kb < DM; kb += 32) {
    __builtin_prefetch(arow + kb + 128, 0, 0);
    c = wmma16(frag_a_f32(arow + kb, g), frag_bt_f32(brow + kb, g), c);
  }
  const int o = o0 + lm;
  const int h = o >> 6, d = o & 63;
  if (p == 2) {                               // V: transposed (B,H,DK,L) layout
#pragma unroll
    for (int r = 0; r < 8; ++r) {
      const int m  = mbase + r + 8 * g;
      const int bI = m >> 11;
      const int l  = m & (L_SEQ - 1);
      Vt[(((size_t)(bI * NH + h) * DK) + d) * L_SEQ + l] = (_Float16)c[r];
    }
  } else {
    _Float16* O = (p == 0) ? Qh : Kh;         // (B,H,L,DK) layout
#pragma unroll
    for (int r = 0; r < 8; ++r) {
      const int m  = mbase + r + 8 * g;
      const int bI = m >> 11;
      const int l  = m & (L_SEQ - 1);
      O[(((size_t)(bI * NH + h) * L_SEQ) + l) * DK + d] = (_Float16)c[r];
    }
  }
}

// ---------------------------------------------------------------------------
// Kernel 2 (FUSED): per block = one (b,h, 16-query strip):
//   phase 1: scores = (Q.K^T)/8 + mask -> LDS (f32, 16x2048 = 128 KB)
//   phase 2: row softmax in LDS; write p_attn (f32, once) + P16 LDS (f16, 64 KB)
//   phase 3: ctx = P16 @ V, all 8 waves (4 d-tiles x 2 K-halves), LDS reduce
// One pass over the 537 MB p_attn buffer instead of four.
// ---------------------------------------------------------------------------
__global__ void __launch_bounds__(256, 1)
k_attn_fused(const _Float16* __restrict__ Qh, const _Float16* __restrict__ Kh,
             const _Float16* __restrict__ Vt, const int* __restrict__ mask,
             float* __restrict__ p_attn, _Float16* __restrict__ ctx2) {
  __shared__ float    S[16 * L_SEQ];     // 128 KB score strip (reused as reduce pad)
  __shared__ _Float16 P16[16 * L_SEQ];   // 64 KB probability strip

  const int lane = threadIdx.x & 31, lm = lane & 15, g = lane >> 4;
  const int wave = threadIdx.x >> 5;
  const int tq = blockIdx.x & 127;            // 128 q strips
  const int bh = blockIdx.x >> 7;             // 32 (b,h)
  const int b  = bh >> 4, h = bh & 15;
  const int q0 = tq * 16;

  // ---- phase 1: scores into LDS --------------------------------------------
  const _Float16* arow = Qh + ((size_t)bh * L_SEQ + q0 + lm) * DK;
  const v16h aq0 = frag_a_f16(arow, g);
  const v16h aq1 = frag_a_f16(arow + 32, g);
  for (int tk = wave; tk < 128; tk += 8) {    // 16 k-tiles per wave
    const _Float16* brow = Kh + ((size_t)bh * L_SEQ + tk * 16 + lm) * DK;
    v8f c = {};
    c = wmma16(aq0, frag_bt_f16(brow, g), c);
    c = wmma16(aq1, frag_bt_f16(brow + 32, g), c);
    const int kpos = tk * 16 + lm;
#pragma unroll
    for (int r = 0; r < 8; ++r) {
      const int qr = r + 8 * g;               // row within strip
      const int mv = mask[((size_t)b * L_SEQ + q0 + qr) * L_SEQ + kpos];
      float s = c[r] * 0.125f;                // 1/TEMPERATURE
      if (mv == 0) s = -1.0e9f;
      S[qr * L_SEQ + kpos] = s;
    }
  }
  __syncthreads();

  // ---- phase 2: softmax (each wave owns 2 full rows) -----------------------
#pragma unroll
  for (int rr = 0; rr < 2; ++rr) {
    const int qr = wave * 2 + rr;
    float* srow = &S[qr * L_SEQ];
    float vmax = -3.0e38f;
    for (int i = lane; i < L_SEQ; i += 32) vmax = fmaxf(vmax, srow[i]);
#pragma unroll
    for (int off = 16; off > 0; off >>= 1) vmax = fmaxf(vmax, __shfl_xor(vmax, off, 32));
    float sum = 0.f;
    for (int i = lane; i < L_SEQ; i += 32) {
      float e = __expf(srow[i] - vmax);
      srow[i] = e;
      sum += e;
    }
#pragma unroll
    for (int off = 16; off > 0; off >>= 1) sum += __shfl_xor(sum, off, 32);
    const float inv = 1.0f / sum;
    float* prow = p_attn + ((size_t)bh * L_SEQ + q0 + qr) * L_SEQ;
    for (int i = lane; i < L_SEQ; i += 32) {
      const float p = srow[i] * inv;
      prow[i] = p;                            // the ONLY global p_attn write
      P16[qr * L_SEQ + i] = (_Float16)p;
    }
  }
  __syncthreads();   // P16 complete; S free for reuse

  // ---- phase 3: ctx = P16 @ V. 8 waves = 4 d-tiles x 2 K-halves ------------
  {
    const int td = wave & 3, kh = wave >> 2;
    const _Float16* vrow = Vt + ((size_t)bh * DK + td * 16 + lm) * L_SEQ;
    v8f c = {};
    const int kb0 = kh * (L_SEQ / 2);
    for (int kb = kb0; kb < kb0 + L_SEQ / 2; kb += 32) {
      const v16h a  = frag_a_f16(&P16[lm * L_SEQ + kb], g);  // ds_load-fed A frag
      const v16h bb = frag_bt_f16(vrow + kb, g);             // contiguous 32B frag
      c = wmma16(a, bb, c);
    }
    float* red = S;   // 4 KB reduction pad carved from the free score LDS
    if (kh == 1) {
#pragma unroll
      for (int r = 0; r < 8; ++r) red[td * 256 + (r + 8 * g) * 16 + lm] = c[r];
    }
    __syncthreads();
    if (kh == 0) {
#pragma unroll
      for (int r = 0; r < 8; ++r) {
        const float v = c[r] + red[td * 256 + (r + 8 * g) * 16 + lm];
        const int q = q0 + r + 8 * g;
        ctx2[((size_t)(b * L_SEQ + q)) * DM + h * DK + td * 16 + lm] = (_Float16)v;
      }
    }
  }
}

// ---------------------------------------------------------------------------
// Kernel 3: out = ctx2 @ Wo^T  (f32 result into d_out[0 .. 4096*1024)).
// ---------------------------------------------------------------------------
__global__ void k_out(const _Float16* __restrict__ ctx2, const float* __restrict__ Wo,
                      float* __restrict__ out) {
  const int lane = threadIdx.x & 31, lm = lane & 15, g = lane >> 4;
  const int wave = threadIdx.x >> 5;
  const int tile = blockIdx.x * 8 + wave;     // 256*64 tiles
  const int tm = tile & 255;
  const int tn = tile >> 8;
  const _Float16* arow = ctx2 + (size_t)(tm * 16 + lm) * DM;
  const float*    brow = Wo   + (size_t)(tn * 16 + lm) * DM;
  v8f c = {};
#pragma unroll 4
  for (int kb = 0; kb < DM; kb += 32)
    c = wmma16(frag_a_f16(arow + kb, g), frag_bt_f32(brow + kb, g), c);
#pragma unroll
  for (int r = 0; r < 8; ++r) {
    const int m = tm * 16 + r + 8 * g;
    out[(size_t)m * DM + tn * 16 + lm] = c[r];
  }
}

// ---------------------------------------------------------------------------
extern "C" void kernel_launch(void* const* d_in, const int* in_sizes, int n_in,
                              void* d_out, int out_size, void* d_ws, size_t ws_size,
                              hipStream_t stream) {
  const float* q_in = (const float*)d_in[0];
  const float* k_in = (const float*)d_in[1];
  const float* v_in = (const float*)d_in[2];
  const int*   mask = (const int*)d_in[3];
  const float* Wq = (const float*)d_in[4];
  const float* Wk = (const float*)d_in[5];
  const float* Wv = (const float*)d_in[6];
  const float* Wo = (const float*)d_in[7];

  float* out    = (float*)d_out;                        // (B, L, DM) f32
  float* p_attn = out + (size_t)M_TOT * DM;             // (B, H, L, L) f32

  const size_t per = (size_t)BH * L_SEQ * DK;           // 4M halfs = 8 MB each
  _Float16* Qh = (_Float16*)d_ws;
  _Float16* Kh = Qh + per;
  _Float16* Vt = Kh + per;                              // (B, H, DK, L) f16
  _Float16* C2 = Vt + per;                              // (B, L, H*DK) f16

  k_qkv_proj  <<<6144, 256, 0, stream>>>(q_in, k_in, v_in, Wq, Wk, Wv, Qh, Kh, Vt);
  k_attn_fused<<<4096, 256, 0, stream>>>(Qh, Kh, Vt, mask, p_attn, C2);
  k_out       <<<2048, 256, 0, stream>>>(C2, Wo, out);
}
